// SupConLoss_57896159150144
// MI455X (gfx1250) — matrix operand order repaired and checked by reading, not compiled, not run on hardware
//
#include <hip/hip_runtime.h>
#include <hip/hip_bf16.h>

typedef __attribute__((ext_vector_type(16))) __bf16 v16bf;
typedef __attribute__((ext_vector_type(8)))  float  v8f;

#define TILE      128
#define KSTEP     32
#define LDA       40      // LDS row stride in bf16 elems (multiple of 8 for b128 align, pads banks)
#define INV_T     (1.0f / 0.07f)
#define EPSV      1e-12f

union FragBF { v16bf v; uint4 q[2]; };

__device__ __forceinline__ unsigned short f2bf(float f) {
    unsigned int x = __float_as_uint(f);
    unsigned int r = x + 0x7fffu + ((x >> 16) & 1u);   // round-to-nearest-even
    return (unsigned short)(r >> 16);
}

// CDNA5 async global->LDS copy, 16B per lane, tracked with ASYNCcnt.
__device__ __forceinline__ void async_load_b128(unsigned lds_off, const void* gptr) {
    asm volatile("global_load_async_to_lds_b128 %0, %1, off"
                 :: "v"(lds_off), "v"(gptr)
                 : "memory");
}
__device__ __forceinline__ void wait_asynccnt0() {
    asm volatile("s_wait_asynccnt 0x0" ::: "memory");
}
__device__ __forceinline__ unsigned lds_offset(const void* p) {
    // generic LDS pointer = {SHARED_BASE aperture, lds_offset[31:0]}
    return (unsigned)(unsigned long long)p;
}

// ---------------------------------------------------------------------------
// Kernel 1: row L2-normalize fp32 -> bf16, and zero per-row accumulators.
// One block (256 threads) per row.
// ---------------------------------------------------------------------------
__global__ void normalize_kernel(const float* __restrict__ feat,
                                 unsigned short* __restrict__ fbf,
                                 float* __restrict__ expsum,
                                 float* __restrict__ msum,
                                 float* __restrict__ cnt,
                                 int D) {
    const int row = blockIdx.x;
    const int tid = threadIdx.x;
    const float* src = feat + (size_t)row * D;
    unsigned short* dst = fbf + (size_t)row * D;

    __shared__ float red[256];
    float ss = 0.0f;
    for (int c = tid * 4; c < D; c += 256 * 4) {
        float4 v = *(const float4*)(src + c);
        ss += v.x * v.x + v.y * v.y + v.z * v.z + v.w * v.w;
    }
    red[tid] = ss;
    __syncthreads();
    for (int s = 128; s > 0; s >>= 1) {
        if (tid < s) red[tid] += red[tid + s];
        __syncthreads();
    }
    const float inv = 1.0f / fmaxf(sqrtf(red[0]), EPSV);

    for (int c = tid * 4; c < D; c += 256 * 4) {
        float4 v = *(const float4*)(src + c);
        unsigned short o0 = f2bf(v.x * inv);
        unsigned short o1 = f2bf(v.y * inv);
        unsigned short o2 = f2bf(v.z * inv);
        unsigned short o3 = f2bf(v.w * inv);
        uint2 pk;
        pk.x = (unsigned int)o0 | ((unsigned int)o1 << 16);
        pk.y = (unsigned int)o2 | ((unsigned int)o3 << 16);
        *(uint2*)(dst + c) = pk;
    }
    if (tid == 0) {
        expsum[row] = 0.0f;
        msum[row]   = 0.0f;
        cnt[row]    = 0.0f;
    }
}

// ---------------------------------------------------------------------------
// Kernel 2: fused tiled bf16 GEMM (f * f^T) with on-the-fly SupCon row
// reductions. Block = 256 threads (8 waves), output tile 128x128.
// Wave w: rows (w&3)*32..+32, cols (w>>2)*64..+64  -> 2x4 WMMA tiles.
// Staging uses CDNA5 async global->LDS (ASYNCcnt) instead of a VGPR bounce.
// ---------------------------------------------------------------------------
__global__ void __launch_bounds__(256)
supcon_gemm(const unsigned short* __restrict__ fbf,
            const int* __restrict__ labels,
            float* __restrict__ expsum,
            float* __restrict__ msum,
            float* __restrict__ cnt,
            int N, int D) {
    __shared__ __align__(16) unsigned short As[TILE * LDA];
    __shared__ __align__(16) unsigned short Bs[TILE * LDA];
    __shared__ int labI[TILE];
    __shared__ int labJ[TILE];

    const int tid   = threadIdx.x;
    const int lane  = tid & 31;
    const int w     = tid >> 5;
    const int ibase = blockIdx.y * TILE;
    const int jbase = blockIdx.x * TILE;
    const int wrow  = (w & 3) * 32;
    const int wcol  = (w >> 2) * 64;

    // staging indices: 512 uint4 per matrix per k-step, 2 per thread
    const int idx0 = tid, idx1 = tid + 256;
    const int rA0 = idx0 >> 2, cA0 = (idx0 & 3) * 8;
    const int rA1 = idx1 >> 2, cA1 = (idx1 & 3) * 8;
    const unsigned short* gA = fbf + (size_t)ibase * D;
    const unsigned short* gB = fbf + (size_t)jbase * D;

    const unsigned asOff0 = lds_offset(&As[rA0 * LDA + cA0]);
    const unsigned asOff1 = lds_offset(&As[rA1 * LDA + cA1]);
    const unsigned bsOff0 = lds_offset(&Bs[rA0 * LDA + cA0]);
    const unsigned bsOff1 = lds_offset(&Bs[rA1 * LDA + cA1]);

    if (tid < TILE) {
        labI[tid] = labels[ibase + tid];
        labJ[tid] = labels[jbase + tid];
    }

    v8f acc[2][4] = {};

    const int m = lane & 15;
    const int g = lane >> 4;

    for (int k0 = 0; k0 < D; k0 += KSTEP) {
        // ---- stage global -> LDS via async copies (16B per lane) ----
        async_load_b128(asOff0, gA + (size_t)rA0 * D + k0 + cA0);
        async_load_b128(asOff1, gA + (size_t)rA1 * D + k0 + cA1);
        async_load_b128(bsOff0, gB + (size_t)rA0 * D + k0 + cA0);
        async_load_b128(bsOff1, gB + (size_t)rA1 * D + k0 + cA1);
        if (k0 + KSTEP < D) {
            // pull next k-tile into GL2 (SYS-scope prefetch feeds the async engine)
            __builtin_prefetch(gA + (size_t)rA0 * D + k0 + KSTEP + cA0, 0, 0);
            __builtin_prefetch(gB + (size_t)rA0 * D + k0 + KSTEP + cA0, 0, 0);
        }
        wait_asynccnt0();
        __syncthreads();

        // ---- build fragments (ISA VGPR layouts) ----
        // A 16x32 bf16: lane m holds row m; VGPRs 0-3 = K g*8..g*8+7, VGPRs 4-7 = K 16+g*8..
        FragBF fa[2];
#pragma unroll
        for (int t = 0; t < 2; ++t) {
            const unsigned short* ap = &As[(wrow + t * 16 + m) * LDA + g * 8];
            fa[t].q[0] = *(const uint4*)(ap);
            fa[t].q[1] = *(const uint4*)(ap + 16);
        }
        // B 32x16 bf16: lane n holds column n; VGPRs 0-3 = K g*8..g*8+7, VGPRs 4-7 = K 16+g*8..
        FragBF fb[4];
#pragma unroll
        for (int nt = 0; nt < 4; ++nt) {
            const unsigned short* bp = &Bs[(wcol + nt * 16 + m) * LDA + g * 8];
            fb[nt].q[0] = *(const uint4*)(bp);
            fb[nt].q[1] = *(const uint4*)(bp + 16);
        }

#pragma unroll
        for (int t = 0; t < 2; ++t)
#pragma unroll
            for (int nt = 0; nt < 4; ++nt)
                acc[t][nt] = __builtin_amdgcn_wmma_f32_16x16x32_bf16(
                    false, fa[t].v, false, fb[nt].v,
                    (short)0, acc[t][nt], false, false);

        __syncthreads();
    }

    // ---- epilogue: per-row reductions ----
    // C layout: VGPR v, lanes 0-15 -> (M=v, N=lane); lanes 16-31 -> (M=v+8, N=lane-16)
#pragma unroll
    for (int t = 0; t < 2; ++t) {
#pragma unroll
        for (int v = 0; v < 8; ++v) {
            const int li = wrow + t * 16 + v + g * 8;
            const int gi = ibase + li;
            const int Li = labI[li];
            float se = 0.0f, sm = 0.0f, sc = 0.0f;
#pragma unroll
            for (int nt = 0; nt < 4; ++nt) {
                const int lj = wcol + nt * 16 + m;
                const int gj = jbase + lj;
                const float sim = acc[t][nt][v] * INV_T;
                if (gi != gj) se += __expf(sim);
                if (Li == labJ[lj]) { sm += sim; sc += 1.0f; }
            }
            // butterfly over the 16 lanes that share this row
#pragma unroll
            for (int off = 1; off < 16; off <<= 1) {
                se += __shfl_xor(se, off);
                sm += __shfl_xor(sm, off);
                sc += __shfl_xor(sc, off);
            }
            if (m == 0) {
                atomicAdd(&expsum[gi], se);
                atomicAdd(&msum[gi],   sm);
                atomicAdd(&cnt[gi],    sc);
            }
        }
    }
}

// ---------------------------------------------------------------------------
// Kernel 3: per-row loss + mean
// ---------------------------------------------------------------------------
__global__ void finalize_kernel(const float* __restrict__ expsum,
                                const float* __restrict__ msum,
                                const float* __restrict__ cnt,
                                float* __restrict__ out, int N) {
    __shared__ float red[256];
    const int tid = threadIdx.x;
    float acc = 0.0f;
    for (int i = tid; i < N; i += 256) {
        const float S  = expsum[i] + EPSV;
        const float c  = cnt[i];
        const float lp = msum[i] - c * __logf(S);
        acc += -lp / (c + EPSV);
    }
    red[tid] = acc;
    __syncthreads();
    for (int s = 128; s > 0; s >>= 1) {
        if (tid < s) red[tid] += red[tid + s];
        __syncthreads();
    }
    if (tid == 0) out[0] = red[0] / (float)N;
}

extern "C" void kernel_launch(void* const* d_in, const int* in_sizes, int n_in,
                              void* d_out, int out_size, void* d_ws, size_t ws_size,
                              hipStream_t stream) {
    const float* feat   = (const float*)d_in[0];
    const int*   labels = (const int*)d_in[1];
    float*       out    = (float*)d_out;

    const int N = in_sizes[1];
    const int D = in_sizes[0] / N;   // 8192, 1024

    char* ws = (char*)d_ws;
    unsigned short* fbf = (unsigned short*)ws;
    size_t featBytes = ((size_t)N * D * sizeof(unsigned short) + 255) & ~(size_t)255;
    float* expsum = (float*)(ws + featBytes);
    float* msum   = expsum + N;
    float* cntp   = msum + N;

    normalize_kernel<<<N, 256, 0, stream>>>(feat, fbf, expsum, msum, cntp, D);

    dim3 grid(N / TILE, N / TILE);
    supcon_gemm<<<grid, 256, 0, stream>>>(fbf, labels, expsum, msum, cntp, N, D);

    finalize_kernel<<<1, 256, 0, stream>>>(expsum, msum, cntp, out, N);
}
